// LongformerSelfAttention_30872224923768
// MI455X (gfx1250) — matrix operand order repaired and checked by reading, not compiled, not run on hardware
//
#include <hip/hip_runtime.h>
#include <hip/hip_bf16.h>

typedef __bf16 bf16_t;
typedef __attribute__((ext_vector_type(16))) __bf16 v16bf;
typedef __attribute__((ext_vector_type(8)))  __bf16 v8bf;
typedef __attribute__((ext_vector_type(4)))  __bf16 v4bf;
typedef __attribute__((ext_vector_type(8)))  float  v8f;

#define BSZ  2
#define SEQ  4096
#define EMB  1024
#define NH   16
#define HD   64
#define WIN  256
#define NBLK 16   // SEQ / WIN

__device__ __forceinline__ v8f wmma_bf16(v16bf a, v16bf b, v8f c) {
  // D = A(16x32) * B(32x16) + C, f32 accumulate
  return __builtin_amdgcn_wmma_f32_16x16x32_bf16(false, a, false, b, (short)0, c,
                                                 false, false);
}

// ---------------------------------------------------------------------------
// Phase 1: one projection GEMM (M=8192, N=1024, K=1024), templated on which
// projection (0=Q,1=K,2=V) so the epilogue is branch-free straight-line code.
// Block = 128x128 tile, 8 waves, each wave 32x64 (8 accumulators), K-step 32.
// LDS tiles are double-buffered and stored directly in WMMA fragment order so
// each fragment load is one contiguous 32B read per lane.
// ---------------------------------------------------------------------------
template <int WID>
__global__ __launch_bounds__(256)
void qkv_proj_kernel(const float* __restrict__ hs,
                     const float* __restrict__ W,
                     const float* __restrict__ bias,
                     bf16_t* __restrict__ dst)
{
  __shared__ bf16_t lA[2][8 * 512];   // 8 m-subtiles, 16x32 bf16, frag order
  __shared__ bf16_t lB[2][8 * 512];   // 8 n-subtiles, 32x16 bf16, frag order

  const int tid = threadIdx.x;
  const int n0  = blockIdx.x * 128;        // 0..1023
  const int m0  = blockIdx.y * 128;        // 0..8191 (B*S rows)

  const int wv = tid >> 5, lane = tid & 31;
  const int wm = wv >> 1, wn = wv & 1;

  v8f acc[2][4];
  const v8f vzero = {0.f, 0.f, 0.f, 0.f, 0.f, 0.f, 0.f, 0.f};
  #pragma unroll
  for (int i = 0; i < 2; i++)
    #pragma unroll
    for (int j = 0; j < 4; j++) acc[i][j] = vzero;

  auto load_tiles = [&](int kk, int buf) {
    // ---- A tile (hidden 128 rows x 32 k): coalesced 128B row segments ----
    #pragma unroll
    for (int i = 0; i < 2; i++) {
      int G = tid + i * 256;               // 512 groups of 8 floats along k
      int r = G >> 2, c0 = (G & 3) * 8;    // 4 consecutive lanes share a row
      const float* src = hs + (size_t)(m0 + r) * EMB + kk + c0;
      float4 f0 = *(const float4*)src;
      float4 f1 = *(const float4*)(src + 4);
      v8bf pk;
      pk[0] = (bf16_t)f0.x; pk[1] = (bf16_t)f0.y;
      pk[2] = (bf16_t)f0.z; pk[3] = (bf16_t)f0.w;
      pk[4] = (bf16_t)f1.x; pk[5] = (bf16_t)f1.y;
      pk[6] = (bf16_t)f1.z; pk[7] = (bf16_t)f1.w;
      // A-frag slot: lane = (r&15)+16*((c>>3)&1), half j = (c&7)+8*(c>>4)
      int sub = r >> 4;
      int L   = (r & 15) + 16 * ((c0 >> 3) & 1);
      int j   = 8 * (c0 >> 4);
      *(v8bf*)&lA[buf][sub * 512 + L * 16 + j] = pk;
    }
    // ---- B tile (weights 32 k x 128 n): loads coalesced along n ----
    #pragma unroll
    for (int i = 0; i < 4; i++) {
      int G = tid + i * 256;               // 1024 groups of 4 elems along k
      int nl = G & 127, c0 = (G >> 7) * 4;
      v4bf pk;
      #pragma unroll
      for (int u = 0; u < 4; u++)
        pk[u] = (bf16_t)W[(size_t)(kk + c0 + u) * EMB + n0 + nl];
      // B-frag slot: lane = (n&15)+16*((c>>3)&1), half j = (c&7)+8*(c>>4)
      int sub = nl >> 4;
      int L   = (nl & 15) + 16 * ((c0 >> 3) & 1);
      int j   = (c0 & 7) + 8 * (c0 >> 4);
      *(v4bf*)&lB[buf][sub * 512 + L * 16 + j] = pk;
    }
  };

  load_tiles(0, 0);
  for (int kk = 0; kk < EMB; kk += 32) {
    const int cur = (kk >> 5) & 1;
    __syncthreads();                        // cur-buf writes visible; prev reads done
    if (kk + 32 < EMB) load_tiles(kk + 32, cur ^ 1);   // overlap with WMMAs

    v16bf a[2], bb[4];
    #pragma unroll
    for (int im = 0; im < 2; im++)
      a[im] = *(const v16bf*)&lA[cur][(wm * 2 + im) * 512 + lane * 16];
    #pragma unroll
    for (int jn = 0; jn < 4; jn++)
      bb[jn] = *(const v16bf*)&lB[cur][(wn * 4 + jn) * 512 + lane * 16];
    #pragma unroll
    for (int im = 0; im < 2; im++)
      #pragma unroll
      for (int jn = 0; jn < 4; jn++)
        acc[im][jn] = wmma_bf16(a[im], bb[jn], acc[im][jn]);
  }

  // ---- epilogue: bias (+ q scale), store bf16; branch-free per WID ----
  const float qscale = (WID == 0) ? 0.125f : 1.0f;   // 1/sqrt(64)
  #pragma unroll
  for (int im = 0; im < 2; im++) {
    #pragma unroll
    for (int jn = 0; jn < 4; jn++) {
      int ncol = n0 + (wn * 4 + jn) * 16 + (lane & 15);
      float bv = bias[ncol];
      #pragma unroll
      for (int e = 0; e < 8; e++) {
        int m = m0 + (wm * 2 + im) * 16 + e + 8 * (lane >> 4);
        float val = (acc[im][jn][e] + bv) * qscale;
        bf16_t h16 = (bf16_t)val;
        if (WID == 2) {
          int bbi = m >> 12, s = m & 4095;
          int h = ncol >> 6, d = ncol & 63;
          dst[(size_t)((bbi * NH + h) * HD + d) * SEQ + s] = h16;  // [B,H,D,S]
        } else {
          dst[(size_t)m * EMB + ncol] = h16;                       // [B*S, H*D]
        }
      }
    }
  }
}

// ---------------------------------------------------------------------------
// Phase 2: banded (sliding-window) attention, flash-style online softmax.
// Block = (head, query-block, batch); 8 waves, wave owns 32 query rows.
// Key band of 768 streamed in steps of 32 keys: 8 score WMMAs + 8 P*V WMMAs.
// ---------------------------------------------------------------------------
__global__ __launch_bounds__(256)
void banded_attn_kernel(const bf16_t* __restrict__ q,
                        const bf16_t* __restrict__ k,
                        const bf16_t* __restrict__ vt,
                        float* __restrict__ out)
{
  __shared__ bf16_t pbuf[8 * 512];   // per-wave 1KB scratch for P transpose
  const int tid = threadIdx.x, wv = tid >> 5, lane = tid & 31;
  const int hb = lane >> 4, ln = lane & 15;
  const int h = blockIdx.x, nblk = blockIdx.y, b = blockIdx.z;

  // ---- Q fragments (A-layout), loaded once: 2 row-tiles x 2 d-chunks ----
  v16bf qa[2][2];
  #pragma unroll
  for (int rt = 0; rt < 2; rt++) {
    int gq = b * SEQ + nblk * WIN + wv * 32 + rt * 16 + ln;
    const bf16_t* qp = q + (size_t)gq * EMB + h * HD;
    #pragma unroll
    for (int dc = 0; dc < 2; dc++) {
      v8bf lo = *(const v8bf*)(qp + dc * 32 + 16 * hb);
      v8bf hi = *(const v8bf*)(qp + dc * 32 + 16 * hb + 8);
      v16bf a;
      #pragma unroll
      for (int u = 0; u < 8; u++) { a[u] = lo[u]; a[u + 8] = hi[u]; }
      qa[rt][dc] = a;
    }
  }

  float mst[2][8], lst[2][8];
  v8f oacc[2][4];
  const v8f vzero = {0.f, 0.f, 0.f, 0.f, 0.f, 0.f, 0.f, 0.f};
  #pragma unroll
  for (int rt = 0; rt < 2; rt++) {
    #pragma unroll
    for (int e = 0; e < 8; e++) { mst[rt][e] = -3.0e38f; lst[rt][e] = 0.f; }
    #pragma unroll
    for (int dv = 0; dv < 4; dv++) oacc[rt][dv] = vzero;
  }

  const int kp0 = nblk * WIN - WIN;      // global pos of band key y=0
  bf16_t* pw = pbuf + wv * 512;

  for (int step = 0; step < 24; step++) {
    const int y0 = step * 32;

    // ---- K fragments (B-layout for S = Q*K^T): 2 key-subtiles x 2 d-chunks
    v16bf kb[2][2];
    #pragma unroll
    for (int kc = 0; kc < 2; kc++) {
      int kpos = kp0 + y0 + kc * 16 + ln;
      int kcl = kpos < 0 ? 0 : (kpos > SEQ - 1 ? SEQ - 1 : kpos);
      const bf16_t* kp_ = k + (size_t)(b * SEQ + kcl) * EMB + h * HD;
      #pragma unroll
      for (int dc = 0; dc < 2; dc++)
        kb[kc][dc] = *(const v16bf*)(kp_ + dc * 32 + 16 * hb);
    }
    // ---- V fragments (B-layout for P*V), contiguous along keys in vT ----
    v16bf vf[4];
    {
      int krun = kp0 + y0 + 16 * hb;               // 16-key run, never straddles
      int kcl = (krun < 0 || krun > SEQ - 16) ? 0 : krun;
      #pragma unroll
      for (int dv = 0; dv < 4; dv++) {
        const bf16_t* vp =
            vt + (size_t)((b * NH + h) * HD + dv * 16 + ln) * SEQ + kcl;
        vf[dv] = *(const v16bf*)vp;
      }
    }

    #pragma unroll
    for (int rt = 0; rt < 2; rt++) {
      v8f s0 = vzero, s1 = vzero;
      #pragma unroll
      for (int dc = 0; dc < 2; dc++) {
        s0 = wmma_bf16(qa[rt][dc], kb[0][dc], s0);
        s1 = wmma_bf16(qa[rt][dc], kb[1][dc], s1);
      }

      // ---- band mask + online softmax (row reductions over 16-lane halves)
      float al[8];
      #pragma unroll
      for (int e = 0; e < 8; e++) {
        int r   = wv * 32 + rt * 16 + e + 8 * hb;   // query pos within block
        int qpg = nblk * WIN + r;
        int k0p = kp0 + y0 + ln;
        int k1p = k0p + 16;
        bool ok0 = (k0p >= 0) && (k0p < SEQ) &&
                   (k0p - qpg <= WIN) && (qpg - k0p <= WIN);
        bool ok1 = (k1p >= 0) && (k1p < SEQ) &&
                   (k1p - qpg <= WIN) && (qpg - k1p <= WIN);
        float x0 = ok0 ? s0[e] : -1.0e30f;
        float x1 = ok1 ? s1[e] : -1.0e30f;
        float vmax = fmaxf(x0, x1);
        vmax = fmaxf(vmax, __shfl_xor(vmax, 1, 32));
        vmax = fmaxf(vmax, __shfl_xor(vmax, 2, 32));
        vmax = fmaxf(vmax, __shfl_xor(vmax, 4, 32));
        vmax = fmaxf(vmax, __shfl_xor(vmax, 8, 32));
        float mo = mst[rt][e];
        float mn = fmaxf(mo, vmax);
        float alpha = __expf(mo - mn);
        float p0 = ok0 ? __expf(x0 - mn) : 0.f;    // exact zero when masked
        float p1 = ok1 ? __expf(x1 - mn) : 0.f;
        float rs = p0 + p1;
        rs += __shfl_xor(rs, 1, 32);
        rs += __shfl_xor(rs, 2, 32);
        rs += __shfl_xor(rs, 4, 32);
        rs += __shfl_xor(rs, 8, 32);
        lst[rt][e] = lst[rt][e] * alpha + rs;
        mst[rt][e] = mn;
        al[e] = alpha;
        s0[e] = p0; s1[e] = p1;
      }
      #pragma unroll
      for (int dv = 0; dv < 4; dv++)
        #pragma unroll
        for (int e = 0; e < 8; e++)
          oacc[rt][dv][e] *= al[e];

      // ---- C-layout P -> A-layout P via wave-private LDS scratch ----
      #pragma unroll
      for (int e = 0; e < 8; e++) {
        int r  = e + 8 * hb;
        int La = r;
        int j  = ln;
        pw[La * 16 + j]     = (bf16_t)s0[e];   // keys c = ln
        pw[(La + 16) * 16 + j] = (bf16_t)s1[e];   // keys c = 16 + ln
      }
      asm volatile("s_wait_dscnt 0" ::: "memory");   // in-wave RAW ordering
      v16bf pf = *(const v16bf*)(pw + lane * 16);

      #pragma unroll
      for (int dv = 0; dv < 4; dv++)
        oacc[rt][dv] = wmma_bf16(pf, vf[dv], oacc[rt][dv]);
    }
  }

  // ---- epilogue: O / l, f32 output [B, S, H*D] ----
  #pragma unroll
  for (int rt = 0; rt < 2; rt++) {
    #pragma unroll
    for (int e = 0; e < 8; e++) {
      int r = wv * 32 + rt * 16 + e + 8 * hb;
      float inv = 1.0f / lst[rt][e];
      size_t row = (size_t)(b * SEQ + nblk * WIN + r) * EMB + h * HD;
      #pragma unroll
      for (int dv = 0; dv < 4; dv++)
        out[row + dv * 16 + ln] = oacc[rt][dv][e] * inv;
    }
  }
}

extern "C" void kernel_launch(void* const* d_in, const int* in_sizes, int n_in,
                              void* d_out, int out_size, void* d_ws, size_t ws_size,
                              hipStream_t stream) {
  (void)in_sizes; (void)n_in; (void)out_size; (void)ws_size;
  const float* hs = (const float*)d_in[0];
  const float* Wq = (const float*)d_in[1];
  const float* bq = (const float*)d_in[2];
  const float* Wk = (const float*)d_in[3];
  const float* bk = (const float*)d_in[4];
  const float* Wv = (const float*)d_in[5];
  const float* bv = (const float*)d_in[6];
  float* out = (float*)d_out;

  bf16_t* qws = (bf16_t*)d_ws;                         // [B*S, E] bf16
  bf16_t* kws = qws + (size_t)BSZ * SEQ * EMB;         // [B*S, E] bf16
  bf16_t* vws = kws + (size_t)BSZ * SEQ * EMB;         // [B, H, D, S] bf16

  dim3 pgrid(EMB / 128, (BSZ * SEQ) / 128);
  qkv_proj_kernel<0><<<pgrid, 256, 0, stream>>>(hs, Wq, bq, qws);
  qkv_proj_kernel<1><<<pgrid, 256, 0, stream>>>(hs, Wk, bk, kws);
  qkv_proj_kernel<2><<<pgrid, 256, 0, stream>>>(hs, Wv, bv, vws);
  banded_attn_kernel<<<dim3(NH, NBLK, BSZ), 256, 0, stream>>>(
      qws, kws, vws, out);
}
